// TERN_62560493634113
// MI455X (gfx1250) — matrix operand order repaired and checked by reading, not compiled, not run on hardware
//
#include <hip/hip_runtime.h>
#include <hip/hip_bf16.h>

// ---------------------------------------------------------------------------
// Problem constants
// ---------------------------------------------------------------------------
#define NB   4
#define NC   64
#define NT   8
#define NX   96
#define NY   96
#define XY   (NX * NY)                 // 9216
#define NN   (NB * XY)                 // 36864  attention batch
#define ROWS (NT * NN)                 // 294912 GEMM rows (t*NN + n)
#define VOL  ((size_t)NC * ROWS)       // 18,874,368 elements per tensor
#define KTOT 576                       // conv0 GEMM K = 64ci * 3 * 3
#define CNT_PER_C ((float)(NB * NT * XY))
#define NEG_SLOPE 0.01f
#define BN_EPS    1e-5f

typedef __attribute__((ext_vector_type(16))) __bf16 bf16x16;
typedef __attribute__((ext_vector_type(8)))  float  f32x8;

#if __has_builtin(__builtin_amdgcn_tensor_load_to_lds) && \
    __has_builtin(__builtin_amdgcn_s_wait_tensorcnt)
#define HAVE_TDM 1
typedef __attribute__((ext_vector_type(4))) unsigned int u32x4;
typedef __attribute__((ext_vector_type(4))) int          i32x4;
typedef __attribute__((ext_vector_type(8))) int          i32x8;
#endif

union PackBf4 {                         // 4 bf16 <-> 8 bytes
  __bf16 h[4];
  uint2  u;
};

// ---------------------------------------------------------------------------
// WMMA fragment loaders (CDNA5 ISA 7.12.2, wave32)
// ---------------------------------------------------------------------------
// A: 16x32 bf16 stored row-major [m][k], row stride lda.
__device__ __forceinline__ bf16x16 frag_a(const __bf16* a, int lda, int lane) {
  const int m  = lane & 15;
  const int hf = lane >> 4;
  bf16x16 r;
#pragma unroll
  for (int g = 0; g < 8; ++g) {
    const int k = ((g < 4) ? 0 : 16) + (hf ? 8 : 0) + 2 * (g & 3);
    r[2 * g]     = a[m * lda + k];
    r[2 * g + 1] = a[m * lda + k + 1];
  }
  return r;
}
// B: 32x16 from conv0 window, lane supplies n = col; elements over k (table order).
__device__ __forceinline__ bf16x16 frag_b_row(const __bf16* w, int kbase, int col, int lane) {
  const int hf = lane >> 4;
  bf16x16 r;
#pragma unroll
  for (int j = 0; j < 16; ++j) {
    const int k  = kbase + ((j < 8) ? (8 * hf + j) : (16 + 8 * hf + j - 8));
    const int ci = k / 9;
    const int r9 = k - ci * 9;
    const int kh = r9 / 3;
    const int kw = r9 - kh * 3;
    r[j] = w[ci * (3 * 98) + kh * 98 + kw + col];
  }
  return r;
}
// B: 32x16 stored transposed [n][k] with row stride ld; lane supplies n, elements over k.
__device__ __forceinline__ bf16x16 frag_bT(const __bf16* bt, int ld, int lane) {
  const int hf = lane >> 4;
  bf16x16 r;
#pragma unroll
  for (int j = 0; j < 16; ++j)
    r[j] = bt[(lane & 15) * ld + ((j < 8) ? (8 * hf + j) : (16 + 8 * hf + j - 8))];
  return r;
}

__device__ __forceinline__ float leaky(float v) {
  return v >= 0.f ? v : NEG_SLOPE * v;
}

__device__ __forceinline__ float block_reduce_256(float v, float* sh) {
  const int tid = threadIdx.x;
  sh[tid] = v;
  __syncthreads();
#pragma unroll
  for (int s = 128; s > 0; s >>= 1) {
    if (tid < s) sh[tid] += sh[tid + s];
    __syncthreads();
  }
  return sh[0];
}

// ---------------------------------------------------------------------------
// K0a: zero BN accumulators
// ---------------------------------------------------------------------------
__global__ void zero_stats_kernel(float* p, int n) {
  int i = blockIdx.x * 256 + threadIdx.x;
  if (i < n) p[i] = 0.f;
}

// ---------------------------------------------------------------------------
// K0b: convert conv0 weights fp32 [64][64*9] -> bf16 [co][k], k-major.
// ---------------------------------------------------------------------------
__global__ void wcvt_kernel(const float* __restrict__ w, __bf16* __restrict__ wbf) {
  const int i = 4 * (blockIdx.x * 256 + threadIdx.x);
  if (i >= NC * KTOT) return;
  const float4 v = *(const float4*)(w + i);
  PackBf4 pk;
  pk.h[0] = (__bf16)v.x; pk.h[1] = (__bf16)v.y;
  pk.h[2] = (__bf16)v.z; pk.h[3] = (__bf16)v.w;
  *(uint2*)(wbf + i) = pk.u;
}

// ---------------------------------------------------------------------------
// K1: conv0 (1,3,3) implicit GEMM, bf16 WMMA.
// grid = B*T*X blocks of 128 threads (4 waves); block = 64co x 96 pixels (row).
// Input window [64ci][3 kh][98 y] staged once (branchless); weights from L2.
// ---------------------------------------------------------------------------
__global__ void conv0_kernel(const float* __restrict__ h,
                             const __bf16* __restrict__ wbf,  // [64][576]
                             const float* __restrict__ bias,
                             float* __restrict__ x0) {
  __shared__ __bf16 win[NC * 3 * 98];     // 37,632 B

  int blk = blockIdx.x;
  const int x = blk % NX; blk /= NX;
  const int t = blk % NT;
  const int b = blk / NT;

  const int tid  = threadIdx.x;
  const int wv   = tid >> 5;
  const int lane = tid & 31;

  const float* hslice = h + (((size_t)b * NC) * NT + t) * XY;

  // Stage input window: branchless clamped loads (no divergent exec loops).
  for (int e = tid; e < NC * 3 * 98; e += 128) {
    const int ci = e / (3 * 98);
    const int r  = e - ci * (3 * 98);
    const int kh = r / 98;
    const int yy = r - kh * 98 - 1;             // -1 .. 96
    const int xx = x + kh - 1;
    const int inb = (xx >= 0) & (xx < NX) & (yy >= 0) & (yy < NY);
    const int xs = min(max(xx, 0), NX - 1);
    const int ys = min(max(yy, 0), NY - 1);
    const float v = hslice[(size_t)ci * NT * XY + xs * NY + ys];
    win[e] = (__bf16)(inb ? v : 0.f);
  }
  __syncthreads();

  f32x8 acc[6] = {};
  const __bf16* arow = wbf + (wv * 16) * KTOT;   // this wave's 16-co A rows

  for (int kk = 0; kk < 18; ++kk) {              // K = 576
    const int kbase = kk * 32;


    const bf16x16 a = frag_a(arow + kbase, KTOT, lane);
#pragma unroll
    for (int jt = 0; jt < 6; ++jt) {
      const bf16x16 bf = frag_b_row(win, kbase, jt * 16 + (lane & 15), lane);
      acc[jt] = __builtin_amdgcn_wmma_f32_16x16x32_bf16(false, a, false, bf,
                                                        (short)0, acc[jt],
                                                        false, false);
    }
  }

  const int hf = lane >> 4, nn = lane & 15;
#pragma unroll
  for (int r = 0; r < 8; ++r) {
    const int co = wv * 16 + r + hf * 8;
    const float bia = bias[co];
    const size_t base = (((size_t)b * NC + co) * NT + t) * XY + x * NY;
#pragma unroll
    for (int jt = 0; jt < 6; ++jt)
      x0[base + jt * 16 + nn] = acc[jt][r] + bia;
  }
}

// ---------------------------------------------------------------------------
// K2: per-channel sum / sum-of-squares (BN training stats). grid (64, 64).
// ---------------------------------------------------------------------------
__global__ void stats_kernel(const float* __restrict__ x,
                             float* __restrict__ sum, float* __restrict__ sq) {
  __shared__ float s1[256];
  __shared__ float s2[256];
  const int c = blockIdx.x;
  const int chunk = blockIdx.y;
  float a = 0.f, q = 0.f;
  const int j0 = chunk * 4608;
  for (int j = j0 + threadIdx.x; j < j0 + 4608; j += 256) {
    const int bn  = j / (NT * XY);
    const int rem = j - bn * (NT * XY);
    const float v = x[((size_t)bn * NC + c) * (NT * XY) + rem];
    a += v;
    q += v * v;
  }
  a = block_reduce_256(a, s1);
  q = block_reduce_256(q, s2);
  if (threadIdx.x == 0) {
    atomicAdd(&sum[c], a);
    atomicAdd(&sq[c], q);
  }
}

__global__ void finalize_stats_kernel(const float* __restrict__ sum,
                                      const float* __restrict__ sq,
                                      float* __restrict__ mean,
                                      float* __restrict__ rstd) {
  const int c = threadIdx.x;
  if (c < NC) {
    const float mu  = sum[c] / CNT_PER_C;
    const float var = sq[c] / CNT_PER_C - mu * mu;
    mean[c] = mu;
    rstd[c] = rsqrtf(var + BN_EPS);
  }
}

// ---------------------------------------------------------------------------
// K3: BN0 + leaky; write fp32 (residual) and bf16 [T,N,C] (attention input).
// ---------------------------------------------------------------------------
__global__ void bn0_apply_kernel(const float* __restrict__ x0,
                                 const float* __restrict__ mean,
                                 const float* __restrict__ rstd,
                                 const float* __restrict__ g,
                                 const float* __restrict__ be,
                                 float* __restrict__ xaf,
                                 __bf16* __restrict__ xab) {
  const size_t i = (size_t)blockIdx.x * 256 + threadIdx.x;
  if (i >= VOL) return;
  const int bc  = (int)(i / (NT * XY));
  const int c   = bc & (NC - 1);
  const int b   = bc >> 6;
  const int rem = (int)(i % (NT * XY));
  const int t   = rem / XY;
  const int p   = rem - t * XY;
  float v = (x0[i] - mean[c]) * rstd[c] * g[c] + be[c];
  v = leaky(v);
  xaf[i] = v;
  const int n = b * XY + p;
  xab[((size_t)t * NN + n) * NC + c] = (__bf16)v;
}

// ---------------------------------------------------------------------------
// K4: qkv = xab @ in_proj_w^T + b   (M=ROWS, K=64, N=192), bf16 WMMA.
// ---------------------------------------------------------------------------
__global__ void qkv_kernel(const __bf16* __restrict__ xab,
                           const float* __restrict__ wq,   // [192][64]
                           const float* __restrict__ bq,
                           __bf16* __restrict__ qkv) {
  __shared__ __bf16 Ws[192 * 64];
  const int tid = threadIdx.x;
  for (int i = tid * 4; i < 192 * 64; i += 256 * 4) {
    const float4 v = *(const float4*)(wq + i);
    PackBf4 pk;
    pk.h[0] = (__bf16)v.x; pk.h[1] = (__bf16)v.y;
    pk.h[2] = (__bf16)v.z; pk.h[3] = (__bf16)v.w;
    *(uint2*)(Ws + i) = pk.u;
  }
  __syncthreads();

  const int wv = tid >> 5, lane = tid & 31;
  const int row0 = blockIdx.x * 128 + wv * 16;
  f32x8 acc[12] = {};
#pragma unroll
  for (int ks = 0; ks < 2; ++ks) {
    const int kbase = ks * 32;
    const bf16x16 a = frag_a(xab + (size_t)row0 * NC + kbase, NC, lane);
#pragma unroll
    for (int ct = 0; ct < 12; ++ct) {
      const bf16x16 bf = frag_bT(Ws + (ct * 16) * 64 + kbase, 64, lane);
      acc[ct] = __builtin_amdgcn_wmma_f32_16x16x32_bf16(false, a, false, bf,
                                                        (short)0, acc[ct],
                                                        false, false);
    }
  }
  const int hf = lane >> 4, nn = lane & 15;
#pragma unroll
  for (int ct = 0; ct < 12; ++ct) {
    const int col = ct * 16 + nn;
    const float bb = bq[col];
#pragma unroll
    for (int r = 0; r < 8; ++r) {
      const int row = row0 + r + hf * 8;
      qkv[(size_t)row * 192 + col] = (__bf16)(acc[ct][r] + bb);
    }
  }
}

// ---------------------------------------------------------------------------
// K5: per-(n,head) attention over T=8 (8x8 softmax core, VALU).
// ---------------------------------------------------------------------------
__global__ void attn_kernel(const __bf16* __restrict__ qkv,
                            __bf16* __restrict__ ob) {
  const int idx = blockIdx.x * 256 + threadIdx.x;
  if (idx >= NN * 8) return;
  const int n = idx >> 3;
  const int hh = idx & 7;
  const float scale = 0.35355339059327379f;  // 1/sqrt(8)

  float kx[64], vx[64];
#pragma unroll
  for (int t = 0; t < 8; ++t) {
    const size_t base = ((size_t)t * NN + n) * 192 + hh * 8;
#pragma unroll
    for (int d = 0; d < 8; ++d) {
      kx[t * 8 + d] = (float)qkv[base + 64 + d];
      vx[t * 8 + d] = (float)qkv[base + 128 + d];
    }
  }
#pragma unroll
  for (int tq = 0; tq < 8; ++tq) {
    const size_t qb = ((size_t)tq * NN + n) * 192 + hh * 8;
    float qv[8];
#pragma unroll
    for (int d = 0; d < 8; ++d) qv[d] = (float)qkv[qb + d] * scale;
    float sc[8];
    float mx = -3.4e38f;
#pragma unroll
    for (int tk = 0; tk < 8; ++tk) {
      float dd = 0.f;
#pragma unroll
      for (int d = 0; d < 8; ++d) dd += qv[d] * kx[tk * 8 + d];
      sc[tk] = dd;
      mx = fmaxf(mx, dd);
    }
    float den = 0.f;
#pragma unroll
    for (int tk = 0; tk < 8; ++tk) {
      sc[tk] = __expf(sc[tk] - mx);
      den += sc[tk];
    }
    const float inv = 1.f / den;
    float ov[8] = {0.f, 0.f, 0.f, 0.f, 0.f, 0.f, 0.f, 0.f};
#pragma unroll
    for (int tk = 0; tk < 8; ++tk) {
      const float wgt = sc[tk] * inv;
#pragma unroll
      for (int d = 0; d < 8; ++d) ov[d] += wgt * vx[tk * 8 + d];
    }
    const size_t obase = ((size_t)tq * NN + n) * NC + hh * 8;
#pragma unroll
    for (int d = 0; d < 8; ++d) ob[obase + d] = (__bf16)ov[d];
  }
}

// ---------------------------------------------------------------------------
// K6: out_proj GEMM (M=ROWS, K=64, N=64) + bias + residual -> NCTHW scatter.
// ---------------------------------------------------------------------------
__global__ void outproj_kernel(const __bf16* __restrict__ ob,
                               const float* __restrict__ wo,  // [64][64]
                               const float* __restrict__ bo,
                               const float* __restrict__ xaf,
                               float* __restrict__ sbuf) {
  __shared__ __bf16 Ws[64 * 64];
  const int tid = threadIdx.x;
  for (int i = tid * 4; i < 64 * 64; i += 256 * 4) {
    const float4 v = *(const float4*)(wo + i);
    PackBf4 pk;
    pk.h[0] = (__bf16)v.x; pk.h[1] = (__bf16)v.y;
    pk.h[2] = (__bf16)v.z; pk.h[3] = (__bf16)v.w;
    *(uint2*)(Ws + i) = pk.u;
  }
  __syncthreads();

  const int wv = tid >> 5, lane = tid & 31;
  const int row0 = blockIdx.x * 128 + wv * 16;
  f32x8 acc[4] = {};
#pragma unroll
  for (int ks = 0; ks < 2; ++ks) {
    const int kbase = ks * 32;
    const bf16x16 a = frag_a(ob + (size_t)row0 * NC + kbase, NC, lane);
#pragma unroll
    for (int ct = 0; ct < 4; ++ct) {
      const bf16x16 bf = frag_bT(Ws + (ct * 16) * 64 + kbase, 64, lane);
      acc[ct] = __builtin_amdgcn_wmma_f32_16x16x32_bf16(false, a, false, bf,
                                                        (short)0, acc[ct],
                                                        false, false);
    }
  }
  const int hf = lane >> 4, nn = lane & 15;
#pragma unroll
  for (int r = 0; r < 8; ++r) {          // decode row ONCE per r
    const int row = row0 + r + hf * 8;
    const int t = row / NN;
    const int n = row - t * NN;
    const int b = n / XY;
    const int p = n - b * XY;
    const size_t ad0 = (((size_t)b * NC) * NT + t) * XY + p;
#pragma unroll
    for (int ct = 0; ct < 4; ++ct) {
      const int c = ct * 16 + nn;
      const size_t ad = ad0 + (size_t)c * (NT * XY);
      sbuf[ad] = acc[ct][r] + bo[c] + xaf[ad];
    }
  }
}

// ---------------------------------------------------------------------------
// K8: BN1 + leaky fused with spatial mean over (X,Y). grid = B*C*T.
// ---------------------------------------------------------------------------
__global__ void bn1_mean_kernel(const float* __restrict__ s,
                                const float* __restrict__ mean,
                                const float* __restrict__ rstd,
                                const float* __restrict__ g,
                                const float* __restrict__ be,
                                float* __restrict__ m) {
  __shared__ float sh[256];
  const int bct = blockIdx.x;
  const int c = (bct >> 3) & (NC - 1);
  const size_t base = (size_t)bct * XY;
  const float mu = mean[c], rs = rstd[c], gg = g[c], bb = be[c];
  float a = 0.f;
  for (int p = threadIdx.x; p < XY; p += 256)
    a += leaky((s[base + p] - mu) * rs * gg + bb);
  a = block_reduce_256(a, sh);
  if (threadIdx.x == 0) m[bct] = a * (1.f / (float)XY);
}

// ---------------------------------------------------------------------------
// K9: 1x1 Conv1d  kern[b][o][t] = sum_c m[b][c][t] * w1[o][c] + b1[o]
// ---------------------------------------------------------------------------
__global__ void conv1_kernel(const float* __restrict__ m,
                             const float* __restrict__ w1,
                             const float* __restrict__ b1,
                             float* __restrict__ kern) {
  const int i = blockIdx.x * 256 + threadIdx.x;
  if (i >= NB * 25 * NT) return;
  const int t = i & (NT - 1);
  const int o = (i >> 3) % 25;
  const int b = i / (25 * NT);
  float a = b1[o];
#pragma unroll
  for (int c = 0; c < NC; ++c)
    a += m[((b * NC) + c) * NT + t] * w1[o * NC + c];
  kern[(b * 25 + o) * NT + t] = a;
}

// ---------------------------------------------------------------------------
// K10: dynamic 5x5 depthwise conv of original h. grid = B*C*T.
// Slice (96x96 fp32, 36 KB) staged to dynamic LDS -- via the Tensor Data
// Mover when the toolchain exposes it, else vectorized float4 staging.
// ---------------------------------------------------------------------------
__global__ void dynconv_kernel(const float* __restrict__ h,
                               const float* __restrict__ kern,
                               float* __restrict__ out) {
  extern __shared__ float tile[];                 // XY floats at LDS offset 0
  const int bct = blockIdx.x;                     // (b*NC + c)*NT + t
  const int t = bct & (NT - 1);
  const int b = bct >> 9;
  const size_t base = (size_t)bct * XY;
  const float* src = h + base;

  float k5[25];
#pragma unroll
  for (int o = 0; o < 25; ++o)
    k5[o] = kern[(b * 25 + o) * NT + t];

#if defined(HAVE_TDM)
  if (threadIdx.x < 32) {                         // one wave issues the DMA
    const unsigned long long ga = (unsigned long long)src;
    u32x4 g0;
    g0[0] = 1u;                                   // count=1, user descriptor
    g0[1] = 0u;                                   // lds_addr = 0 (dynamic LDS)
    g0[2] = (unsigned)(ga & 0xffffffffull);       // global_addr[31:0]
    g0[3] = (unsigned)((ga >> 32) & 0x01ffffffull) | 0x80000000u; // type=2
    i32x8 g1;
    g1[0] = 0x20000;                              // data_size=4B
    g1[1] = (int)((XY & 0xffff) << 16);           // tensor_dim0 lo16
    g1[2] = (int)((XY >> 16) | (1u << 16));       // dim0 hi | tensor_dim1=1
    g1[3] = (int)((unsigned)XY << 16);            // tile_dim0 = 9216
    g1[4] = 1;                                    // tile_dim1 = 1
    g1[5] = XY;                                   // tensor_dim0_stride
    g1[6] = 0;
    g1[7] = 0;
    i32x4 gz4 = {0, 0, 0, 0};
    i32x8 gz8 = {0, 0, 0, 0, 0, 0, 0, 0};
    __builtin_amdgcn_tensor_load_to_lds(g0, g1, gz4, gz4, gz8, 0);
  }
  __builtin_amdgcn_s_wait_tensorcnt((short)0);
  __syncthreads();
#else
  for (int i = threadIdx.x * 4; i < XY; i += 256 * 4) {
    *(float4*)(tile + i) = *(const float4*)(src + i);
  }
  __syncthreads();
#endif

  for (int p = threadIdx.x; p < XY; p += 256) {
    const int xx = p / NY;
    const int yy = p - xx * NY;
    float a = 0.f;
#pragma unroll
    for (int i = 0; i < 5; ++i) {
      const int x2 = xx + i - 2;
      if (x2 < 0 || x2 >= NX) continue;
#pragma unroll
      for (int j = 0; j < 5; ++j) {
        const int y2 = yy + j - 2;
        if (y2 < 0 || y2 >= NY) continue;
        a += tile[x2 * NY + y2] * k5[i * 5 + j];
      }
    }
    out[base + p] = a;
  }
}

// ---------------------------------------------------------------------------
// Host: workspace carving + launch sequence (all on `stream`, graph-safe)
// ---------------------------------------------------------------------------
extern "C" void kernel_launch(void* const* d_in, const int* in_sizes, int n_in,
                              void* d_out, int out_size, void* d_ws, size_t ws_size,
                              hipStream_t stream) {
  (void)in_sizes; (void)n_in; (void)out_size; (void)ws_size;
  const float* h        = (const float*)d_in[0];
  const float* conv0_w  = (const float*)d_in[1];
  const float* conv0_b  = (const float*)d_in[2];
  const float* bn0_g    = (const float*)d_in[3];
  const float* bn0_b    = (const float*)d_in[4];
  const float* bn1_g    = (const float*)d_in[5];
  const float* bn1_b    = (const float*)d_in[6];
  const float* inproj_w = (const float*)d_in[7];
  const float* inproj_b = (const float*)d_in[8];
  const float* outp_w   = (const float*)d_in[9];
  const float* outp_b   = (const float*)d_in[10];
  const float* conv1_w  = (const float*)d_in[11];
  const float* conv1_b  = (const float*)d_in[12];
  float* out = (float*)d_out;

  char* ws = (char*)d_ws;
  size_t off = 0;
  auto carve = [&](size_t bytes) -> char* {
    char* p = ws + off;
    off = (off + bytes + 255) & ~(size_t)255;
    return p;
  };
  float*  x0f  = (float*)carve(VOL * 4);                   // conv0 out / residual sum
  float*  xaf  = (float*)carve(VOL * 4);                   // BN0+leaky fp32
  __bf16* xab  = (__bf16*)carve((size_t)ROWS * NC * 2);    // [T,N,C] bf16
  __bf16* qkv  = (__bf16*)carve((size_t)ROWS * 192 * 2);
  __bf16* ob   = (__bf16*)carve((size_t)ROWS * NC * 2);
  __bf16* wbf  = (__bf16*)carve((size_t)NC * KTOT * 2);    // conv0 weights bf16
  float*  sum0 = (float*)carve(NC * 4);
  float*  sq0  = (float*)carve(NC * 4);
  float*  mu0  = (float*)carve(NC * 4);
  float*  rs0  = (float*)carve(NC * 4);
  float*  sum1 = (float*)carve(NC * 4);
  float*  sq1  = (float*)carve(NC * 4);
  float*  mu1  = (float*)carve(NC * 4);
  float*  rs1  = (float*)carve(NC * 4);
  float*  mbuf = (float*)carve((size_t)NB * NC * NT * 4);
  float*  kern = (float*)carve((size_t)NB * 25 * NT * 4);

  zero_stats_kernel<<<2, 256, 0, stream>>>(sum0, 8 * NC);
  wcvt_kernel<<<(NC * KTOT) / 1024, 256, 0, stream>>>(conv0_w, wbf);
  conv0_kernel<<<NB * NT * NX, 128, 0, stream>>>(h, wbf, conv0_b, x0f);
  stats_kernel<<<dim3(NC, 64), 256, 0, stream>>>(x0f, sum0, sq0);
  finalize_stats_kernel<<<1, 64, 0, stream>>>(sum0, sq0, mu0, rs0);
  bn0_apply_kernel<<<(unsigned)(VOL / 256), 256, 0, stream>>>(
      x0f, mu0, rs0, bn0_g, bn0_b, xaf, xab);
  qkv_kernel<<<ROWS / 128, 256, 0, stream>>>(xab, inproj_w, inproj_b, qkv);
  attn_kernel<<<(NN * 8) / 256, 256, 0, stream>>>(qkv, ob);
  outproj_kernel<<<ROWS / 128, 256, 0, stream>>>(ob, outp_w, outp_b, xaf, x0f);
  stats_kernel<<<dim3(NC, 64), 256, 0, stream>>>(x0f, sum1, sq1);
  finalize_stats_kernel<<<1, 64, 0, stream>>>(sum1, sq1, mu1, rs1);
  bn1_mean_kernel<<<NB * NC * NT, 256, 0, stream>>>(x0f, mu1, rs1, bn1_g, bn1_b, mbuf);
  conv1_kernel<<<(NB * 25 * NT + 255) / 256, 256, 0, stream>>>(mbuf, conv1_w, conv1_b, kern);
  dynconv_kernel<<<NB * NC * NT, 256, XY * sizeof(float), stream>>>(h, kern, out);
}
